// PhysMamba_SSSD_65231963292042
// MI455X (gfx1250) — matrix operand order repaired and checked by reading, not compiled, not run on hardware
//
#include <hip/hip_runtime.h>
#include <hip/hip_bf16.h>
#include <math.h>

// ---------------- problem constants ----------------
#define B_ 4
#define L_ 1024
#define D_ 256
#define H_ 16
#define HD_ 16
#define DI_ 512
#define S_ 64
#define KC_ 4
#define DTR_ 16
#define DBCW_ (DTR_ + 2 * S_)   // 144

typedef __attribute__((ext_vector_type(16))) __bf16 v16bf;
typedef __attribute__((ext_vector_type(8)))  float  v8f;
typedef __attribute__((ext_vector_type(4)))  float  f32x4;

__device__ inline float fast_rcp(float x) { return __builtin_amdgcn_rcpf(x); }
__device__ inline float fast_sigmoid(float x) { return fast_rcp(1.f + __expf(-x)); }
__device__ inline float fast_silu(float x) { return x * fast_sigmoid(x); }
// branch-free numerically-stable softplus: max(v,0) + log(1 + exp(-|v|))
__device__ inline float fast_softplus(float x) {
  return fmaxf(x, 0.f) + __logf(1.f + __expf(-fabsf(x)));
}

// ================= branch-free WMMA fragment loaders (fp32 -> bf16) ============
// A-matrix 16x32 bf16 (ISA 7.12.2): lane r=l&15 is row M; kb=(l>>4)*8;
// elements 0..7 -> K=kb+e ; elements 8..15 -> K=16+kb+e.
// All call sites guarantee 16B alignment (ld % 4 == 0, base 16B aligned).
__device__ inline v16bf fragA_full(const float* base, int ld, int lane) {
  int r = lane & 15, kb = (lane >> 4) * 8;
  const f32x4* p = (const f32x4*)(base + (size_t)r * ld + kb);
  f32x4 x0 = p[0], x1 = p[1], x2 = p[4], x3 = p[5];   // +0,+4,+16,+20 floats
  v16bf f;
#pragma unroll
  for (int e = 0; e < 4; ++e) {
    f[e]      = (__bf16)x0[e];
    f[e + 4]  = (__bf16)x1[e];
    f[e + 8]  = (__bf16)x2[e];
    f[e + 12] = (__bf16)x3[e];
  }
  return f;
}

// A tile with only K=0..15 valid (upper half compile-time zero).
__device__ inline v16bf fragA_k16(const float* base, int ld, int lane) {
  int r = lane & 15;
  float m = (lane & 16) ? 0.f : 1.f;
  const f32x4* p = (const f32x4*)(base + (size_t)r * ld);
  f32x4 x0 = p[0], x1 = p[1], x2 = p[2], x3 = p[3];
  v16bf f;
#pragma unroll
  for (int e = 0; e < 4; ++e) {
    f[e]      = (__bf16)(x0[e] * m);
    f[e + 4]  = (__bf16)(x1[e] * m);
    f[e + 8]  = (__bf16)(x2[e] * m);
    f[e + 12] = (__bf16)(x3[e] * m);
  }
  return f;
}

// B-matrix 32x16 bf16: lane n=l&15 is column N; kb=(l>>4)*8;
// elements 0..7 -> K=kb+e ; elements 8..15 -> K=16+kb+e.
__device__ inline v16bf fragB_full(const float* base, int ld, int lane) {
  int n = lane & 15, kb = (lane >> 4) * 8;
  const f32x4* p = (const f32x4*)(base + (size_t)n * ld + kb);
  f32x4 x0 = p[0], x1 = p[1], x2 = p[4], x3 = p[5];
  v16bf f;
#pragma unroll
  for (int e = 0; e < 4; ++e) {
    f[e]      = (__bf16)x0[e];
    f[e + 4]  = (__bf16)x1[e];
    f[e + 8]  = (__bf16)x2[e];
    f[e + 12] = (__bf16)x3[e];
  }
  return f;
}

// B tile with only K=0..15 valid: upper lane-group masked to zero; all lanes
// load the (in-bounds) K=0..15 span.
__device__ inline v16bf fragB_k16(const float* base, int ld, int lane) {
  int n = lane & 15;
  float m = (lane & 16) ? 0.f : 1.f;
  const f32x4* p = (const f32x4*)(base + (size_t)n * ld);
  f32x4 x0 = p[0], x1 = p[1], x2 = p[2], x3 = p[3];
  v16bf f;
#pragma unroll
  for (int e = 0; e < 4; ++e) {
    f[e]      = (__bf16)(x0[e] * m);
    f[e + 4]  = (__bf16)(x1[e] * m);
    f[e + 8]  = (__bf16)(x2[e] * m);
    f[e + 12] = (__bf16)(x3[e] * m);
  }
  return f;
}

// Column-major (K-major) B tile, K=0..15 valid: element (k,n) at base[k*ld+n].
__device__ inline v16bf fragB_col16(const float* base, int ld, int lane) {
  int n = lane & 15;
  float m = (lane & 16) ? 0.f : 1.f;
  v16bf f;
#pragma unroll
  for (int e = 0; e < 16; ++e)
    f[e] = (__bf16)(base[(size_t)e * ld + n] * m);
  return f;
}

// ================= generic GEMM: C = A(MxK, lda) @ W(NxK)^T + bias =============
// Requires K % 32 == 0.  ACT: 0=none, 1=sigmoid.  8 waves/WG -> 64(M) x 32(N).
template <int ACT, bool HASBIAS>
__global__ __launch_bounds__(256) void gemm_kernel(
    const float* __restrict__ A, int lda,
    const float* __restrict__ W,
    const float* __restrict__ bias,
    float* __restrict__ C, int ldc,
    int M, int N, int K) {
  int wave = threadIdx.x >> 5, lane = threadIdx.x & 31;
  int m0 = blockIdx.y * 64 + (wave >> 1) * 16;
  int n0 = blockIdx.x * 32 + (wave & 1) * 16;
  if (m0 >= M || n0 >= N) return;   // wave-uniform guard (EXEC stays all-ones)
  v8f acc = {};
  for (int k0 = 0; k0 < K; k0 += 32) {
    v16bf a = fragA_full(A + (size_t)m0 * lda + k0, lda, lane);
    v16bf b = fragB_full(W + (size_t)n0 * K + k0, K, lane);
    acc = __builtin_amdgcn_wmma_f32_16x16x32_bf16(false, a, false, b,
                                                  (short)0, acc, false, false);
  }
  int col = n0 + (lane & 15);
  int r0  = m0 + 8 * (lane >> 4);
  float bv = HASBIAS ? bias[col] : 0.f;
#pragma unroll
  for (int j = 0; j < 8; ++j) {
    float v = acc[j] + bv;
    if (ACT == 1) v = fast_sigmoid(v);
    C[(size_t)(r0 + j) * ldc + col] = v;
  }
}

// ============ dt projection: dt = softplus(dbc[:, :16] @ dt_w^T + dt_b) ========
__global__ __launch_bounds__(256) void dt16_kernel(
    const float* __restrict__ dbc, const float* __restrict__ Wt,
    const float* __restrict__ bias, float* __restrict__ C) {
  int wave = threadIdx.x >> 5, lane = threadIdx.x & 31;
  int m0 = blockIdx.y * 64 + (wave >> 1) * 16;
  int n0 = blockIdx.x * 32 + (wave & 1) * 16;
  v16bf a = fragA_k16(dbc + (size_t)m0 * DBCW_, DBCW_, lane);
  v16bf b = fragB_k16(Wt + (size_t)n0 * DTR_, DTR_, lane);
  v8f acc = {};
  acc = __builtin_amdgcn_wmma_f32_16x16x32_bf16(false, a, false, b,
                                                (short)0, acc, false, false);
  int col = n0 + (lane & 15);
  int r0  = m0 + 8 * (lane >> 4);
  float bv = bias[col];
#pragma unroll
  for (int j = 0; j < 8; ++j)
    C[(size_t)(r0 + j) * DI_ + col] = fast_softplus(acc[j] + bv);
}

// ================= LayerNorm: one 256-thread block per token ===================
__global__ __launch_bounds__(256) void ln_kernel(
    const float* __restrict__ x, const float* __restrict__ g,
    const float* __restrict__ bta, float* __restrict__ xn) {
  __shared__ float red[256];
  int t = blockIdx.x, i = threadIdx.x;
  float v = x[(size_t)t * D_ + i];
  red[i] = v; __syncthreads();
  for (int s = 128; s > 0; s >>= 1) { if (i < s) red[i] += red[i + s]; __syncthreads(); }
  float mu = red[0] * (1.f / D_);
  __syncthreads();
  float dlt = v - mu;
  red[i] = dlt * dlt; __syncthreads();
  for (int s = 128; s > 0; s >>= 1) { if (i < s) red[i] += red[i + s]; __syncthreads(); }
  float var = red[0] * (1.f / D_);
  xn[(size_t)t * D_ + i] = dlt * rsqrtf(var + 1e-5f) * g[i] + bta[i];
}

// ================= avg-pool over time (stride=kernel=s) ========================
__global__ void pool_kernel(const float* __restrict__ xn, float* __restrict__ xd,
                            int s, int Ls) {
  int idx = blockIdx.x * blockDim.x + threadIdx.x;
  int tot = B_ * Ls * D_;
  if (idx >= tot) return;
  int dd = idx % D_, t = (idx / D_) % Ls, b = idx / (D_ * Ls);
  const float* src = xn + ((size_t)(b * L_ + t * s)) * D_ + dd;
  float acc = 0.f;
  for (int j = 0; j < s; ++j) acc += src[(size_t)j * D_];
  xd[idx] = acc / (float)s;
}

// ================= linear interp up-sample into concat buffer ==================
__global__ void interp_kernel(const float* __restrict__ qd, float* __restrict__ q4,
                              int Ls, int colofs) {
  int idx = blockIdx.x * blockDim.x + threadIdx.x;
  int tot = B_ * L_ * D_;
  if (idx >= tot) return;
  int dd = idx % D_, l = (idx / D_) % L_, b = idx / (D_ * L_);
  float pos = (l + 0.5f) * ((float)Ls / (float)L_) - 0.5f;
  pos = fminf(fmaxf(pos, 0.f), (float)(Ls - 1));
  int lo = (int)floorf(pos);
  int hi = (lo + 1 < Ls) ? lo + 1 : Ls - 1;
  float w = pos - (float)lo;
  float vlo = qd[((size_t)(b * Ls + lo)) * D_ + dd];
  float vhi = qd[((size_t)(b * Ls + hi)) * D_ + dd];
  q4[((size_t)(b * L_ + l)) * (4 * D_) + colofs + dd] = vlo * (1.f - w) + vhi * w;
}

// ================= causal depthwise conv(KC=4) + SiLU ==========================
__global__ void conv_kernel(const float* __restrict__ xz, const float* __restrict__ cw,
                            const float* __restrict__ cb, float* __restrict__ xc) {
  int idx = blockIdx.x * blockDim.x + threadIdx.x;
  int tot = B_ * L_ * DI_;
  if (idx >= tot) return;
  int dd = idx % DI_, l = (idx / DI_) % L_, b = idx / (DI_ * L_);
  const float* xm = xz + ((size_t)b * L_) * (2 * DI_) + dd; // xm = cols [0,DI)
  float acc = cb[dd];
#pragma unroll
  for (int kk = 0; kk < KC_; ++kk) {
    int tl = l + kk - (KC_ - 1);
    if (tl >= 0) acc += xm[(size_t)tl * (2 * DI_)] * cw[dd * KC_ + kk];
  }
  xc[idx] = fast_silu(acc);
}

// ================= selective scan: 1 wave per (b,d); 2 states/lane =============
__global__ __launch_bounds__(256) void scan_kernel(
    const float* __restrict__ xc, const float* __restrict__ dt,
    const float* __restrict__ dbc, const float* __restrict__ A_log,
    float* __restrict__ ys) {
  int wave = threadIdx.x >> 5, lane = threadIdx.x & 31;
  int d = blockIdx.x * 8 + wave;
  int b = blockIdx.y;
  float A0 = -__expf(A_log[d * S_ + lane]);
  float A1 = -__expf(A_log[d * S_ + lane + 32]);
  float h0 = 0.f, h1 = 0.f;
  const float* xcb  = xc  + ((size_t)b * L_) * DI_ + d;
  const float* dtb  = dt  + ((size_t)b * L_) * DI_ + d;
  const float* dbcb = dbc + ((size_t)b * L_) * DBCW_;
  float* ysb = ys + ((size_t)b * L_) * DI_ + d;
  for (int t = 0; t < L_; ++t) {
    float dtt = dtb[(size_t)t * DI_];
    float xt  = xcb[(size_t)t * DI_];
    const float* row = dbcb + (size_t)t * DBCW_;
    float Bt0 = row[DTR_ + lane],      Bt1 = row[DTR_ + lane + 32];
    float Ct0 = row[DTR_ + S_ + lane], Ct1 = row[DTR_ + S_ + lane + 32];
    h0 = __expf(dtt * A0) * h0 + dtt * xt * Bt0;
    h1 = __expf(dtt * A1) * h1 + dtt * xt * Bt1;
    float p = h0 * Ct0 + h1 * Ct1;
#pragma unroll
    for (int off = 16; off > 0; off >>= 1) p += __shfl_xor(p, off, 32);
    if (lane == 0) ysb[(size_t)t * DI_] = p;
  }
}

// ================= y = (ys + xc*D) * silu(z), in place into ys =================
__global__ void ygate_kernel(float* __restrict__ ys, const float* __restrict__ xc,
                             const float* __restrict__ Dssm, const float* __restrict__ xz) {
  int idx = blockIdx.x * blockDim.x + threadIdx.x;
  int tot = B_ * L_ * DI_;
  if (idx >= tot) return;
  int dd = idx % DI_;
  int row = idx / DI_;
  float z = xz[(size_t)row * (2 * DI_) + DI_ + dd];
  float y = ys[idx] + xc[idx] * Dssm[dd];
  ys[idx] = y * fast_silu(z);
}

// ================= decay-masked attention (WMMA x2 per k-tile) =================
__global__ __launch_bounds__(256) void attn_kernel(
    const float* __restrict__ q, const float* __restrict__ kbuf,
    const float* __restrict__ vbuf, const float* __restrict__ log_tau,
    float* __restrict__ attn) {
  __shared__ float sw[8][256];
  int wave = threadIdx.x >> 5, lane = threadIdx.x & 31;
  int task = blockIdx.x * 8 + wave;           // b*H*(L/16) tasks
  int qt = task & 63;
  int h  = (task >> 6) & 15;
  int b  = task >> 10;
  int q0 = qt * 16;
  int hi = lane >> 4, ln = lane & 15;
  float tau = __expf(log_tau[h]);
  float itau = fast_rcp(tau);
  float rr  = __expf(-itau);
  float irm = fast_rcp(1.f - rr);
  // closed-form mask row-sum: denom(q) = (L-q) + r*(1-r^q)/(1-r); r^q = exp(-q/tau)
  float rdnm[8];
#pragma unroll
  for (int j = 0; j < 8; ++j) {
    int qq = q0 + 8 * hi + j;
    float rq = __expf(-(float)qq * itau);
    float dsum = (float)(L_ - qq) + rr * (1.f - rq) * irm;
    rdnm[j] = 0.25f * fast_rcp(fmaxf(dsum, 1e-6f));   // fold in 1/sqrt(HD)
  }
  v16bf qf = fragA_k16(q + ((size_t)(b * L_ + q0)) * D_ + h * HD_, D_, lane);
  v8f acc = {};
  float* swp = sw[wave];
  for (int kt = 0; kt < L_ / 16; ++kt) {
    int k0 = kt * 16;
    v16bf kf = fragB_k16(kbuf + ((size_t)(b * L_ + k0)) * D_ + h * HD_, D_, lane);
    v8f s = {};
    s = __builtin_amdgcn_wmma_f32_16x16x32_bf16(false, qf, false, kf,
                                                (short)0, s, false, false);
#pragma unroll
    for (int j = 0; j < 8; ++j) {
      int qq = q0 + 8 * hi + j;
      int kk = k0 + ln;
      float dist = (qq > kk) ? (float)(qq - kk) : 0.f;
      swp[(8 * hi + j) * 16 + ln] = s[j] * __expf(-dist * itau) * rdnm[j];
    }
    asm volatile("s_wait_dscnt 0x0" ::: "memory"); // stores landed before re-read
    v16bf af = fragA_k16(swp, 16, lane);           // C-frag -> A-frag via LDS
    v16bf vf = fragB_col16(vbuf + ((size_t)(b * L_ + k0)) * D_ + h * HD_, D_, lane);
    acc = __builtin_amdgcn_wmma_f32_16x16x32_bf16(false, af, false, vf,
                                                  (short)0, acc, false, false);
  }
#pragma unroll
  for (int j = 0; j < 8; ++j)
    attn[((size_t)(b * L_ + q0 + 8 * hi + j)) * D_ + h * HD_ + ln] = acc[j];
}

// ================= final gate fuse + residual ==================================
__global__ void fuse_kernel(const float* __restrict__ x, const float* __restrict__ gate,
                            const float* __restrict__ cat2, float* __restrict__ out) {
  int idx = blockIdx.x * blockDim.x + threadIdx.x;
  int tot = B_ * L_ * D_;
  if (idx >= tot) return;
  int dd = idx % D_;
  int row = idx / D_;
  float g   = gate[idx];
  float ssm = cat2[(size_t)row * (2 * D_) + dd];
  float ssd = cat2[(size_t)row * (2 * D_) + D_ + dd];
  out[idx] = x[idx] + g * ssm + (1.f - g) * ssd;
}

// ===================== host side ===============================================
template <int ACT, bool HASBIAS>
static inline void launch_gemm(const float* A, int lda, const float* W,
                               const float* bias, float* C, int ldc,
                               int M, int N, int K, hipStream_t s) {
  dim3 g((N + 31) / 32, (M + 63) / 64);
  gemm_kernel<ACT, HASBIAS><<<g, 256, 0, s>>>(A, lda, W, bias, C, ldc, M, N, K);
}

extern "C" void kernel_launch(void* const* d_in, const int* in_sizes, int n_in,
                              void* d_out, int out_size, void* d_ws, size_t ws_size,
                              hipStream_t stream) {
  const float* x        = (const float*)d_in[0];
  const float* ln_g     = (const float*)d_in[1];
  const float* ln_b     = (const float*)d_in[2];
  const float* mq_w     = (const float*)d_in[3];   // (4,D,D)
  const float* mq_b     = (const float*)d_in[4];   // (4,D)
  const float* fuse_w   = (const float*)d_in[5];   // (D,4D)
  const float* fuse_b   = (const float*)d_in[6];
  const float* in_proj_w= (const float*)d_in[7];   // (2DI,D)
  const float* conv_w   = (const float*)d_in[8];   // (DI,KC)
  const float* conv_b   = (const float*)d_in[9];
  const float* x_proj_w = (const float*)d_in[10];  // (DTR+2S,DI)
  const float* dt_w     = (const float*)d_in[11];  // (DI,DTR)
  const float* dt_b     = (const float*)d_in[12];
  const float* A_log    = (const float*)d_in[13];  // (DI,S)
  const float* D_ssm    = (const float*)d_in[14];
  const float* out_w    = (const float*)d_in[15];  // (D,DI)
  const float* k_w      = (const float*)d_in[16];
  const float* k_b      = (const float*)d_in[17];
  const float* v_w      = (const float*)d_in[18];
  const float* v_b      = (const float*)d_in[19];
  const float* o_w      = (const float*)d_in[20];
  const float* o_b      = (const float*)d_in[21];
  const float* log_tau  = (const float*)d_in[22];
  const float* gate_w   = (const float*)d_in[23];  // (D,2D)
  const float* gate_b   = (const float*)d_in[24];
  float* out = (float*)d_out;

  const int M = B_ * L_;                 // 4096 tokens
  float* w = (float*)d_ws;
  size_t off = 0;
  auto carve = [&](size_t n) { float* p = w + off; off += n; return p; };

  float* xn   = carve((size_t)M * D_);
  float* q4   = carve((size_t)M * 4 * D_);      // later reused as xz
  float* xp   = carve((size_t)B_ * 512 * D_);   // pooled input (max Ls=512)
  float* qd   = carve((size_t)B_ * 512 * D_);   // per-scale projection
  float* qbuf = carve((size_t)M * D_);
  float* xc   = carve((size_t)M * DI_);
  float* dbc  = carve((size_t)M * DBCW_);
  float* dtb  = carve((size_t)M * DI_);
  float* ysb  = carve((size_t)M * DI_);
  float* kbuf = carve((size_t)M * D_);
  float* vbuf = carve((size_t)M * D_);
  float* attnb= carve((size_t)M * D_);
  float* cat2 = carve((size_t)M * 2 * D_);
  float* gateb= carve((size_t)M * D_);
  float* xz   = q4;                              // alias: consumed before reuse
  (void)ws_size; (void)n_in; (void)in_sizes; (void)out_size;

  auto eblocks = [](int n) { return (n + 255) / 256; };

  // ---- LayerNorm ----
  ln_kernel<<<M, 256, 0, stream>>>(x, ln_g, ln_b, xn);

  // ---- MultiScaleQuery ----
  launch_gemm<0, true>(xn, D_, mq_w, mq_b, q4, 4 * D_, M, D_, D_, stream); // scale 1 -> cols [0,D)
  const int scales[3] = {2, 4, 8};
  for (int i = 0; i < 3; ++i) {
    int s = scales[i], Ls = L_ / s;
    pool_kernel<<<eblocks(B_ * Ls * D_), 256, 0, stream>>>(xn, xp, s, Ls);
    launch_gemm<0, true>(xp, D_, mq_w + (size_t)(i + 1) * D_ * D_, mq_b + (i + 1) * D_,
                         qd, D_, B_ * Ls, D_, D_, stream);
    interp_kernel<<<eblocks(M * D_), 256, 0, stream>>>(qd, q4, Ls, (i + 1) * D_);
  }
  launch_gemm<0, true>(q4, 4 * D_, fuse_w, fuse_b, qbuf, D_, M, D_, 4 * D_, stream);

  // ---- Mamba ----
  launch_gemm<0, false>(xn, D_, in_proj_w, nullptr, xz, 2 * DI_, M, 2 * DI_, D_, stream);
  conv_kernel<<<eblocks(M * DI_), 256, 0, stream>>>(xz, conv_w, conv_b, xc);
  launch_gemm<0, false>(xc, DI_, x_proj_w, nullptr, dbc, DBCW_, M, DBCW_, DI_, stream);
  dt16_kernel<<<dim3(DI_ / 32, M / 64), 256, 0, stream>>>(dbc, dt_w, dt_b, dtb);
  scan_kernel<<<dim3(DI_ / 8, B_), 256, 0, stream>>>(xc, dtb, dbc, A_log, ysb);
  ygate_kernel<<<eblocks(M * DI_), 256, 0, stream>>>(ysb, xc, D_ssm, xz);
  launch_gemm<0, false>(ysb, DI_, out_w, nullptr, cat2, 2 * D_, M, D_, DI_, stream); // ssm cols [0,D)

  // ---- StructuredSSD ----
  launch_gemm<0, true>(xn, D_, k_w, k_b, kbuf, D_, M, D_, D_, stream);
  launch_gemm<0, true>(xn, D_, v_w, v_b, vbuf, D_, M, D_, D_, stream);
  attn_kernel<<<(B_ * H_ * (L_ / 16)) / 8, 256, 0, stream>>>(qbuf, kbuf, vbuf, log_tau, attnb);
  launch_gemm<0, true>(attnb, D_, o_w, o_b, cat2 + D_, 2 * D_, M, D_, D_, stream); // ssd cols [D,2D)

  // ---- gate + residual ----
  launch_gemm<1, true>(cat2, 2 * D_, gate_w, gate_b, gateb, D_, M, D_, 2 * D_, stream); // sigmoid
  fuse_kernel<<<eblocks(M * D_), 256, 0, stream>>>(x, gateb, cat2, out);
}
